// EdgePointGNN_18485539242274
// MI455X (gfx1250) — matrix operand order repaired and checked by reading, not compiled, not run on hardware
//
#include <hip/hip_runtime.h>
#include <hip/hip_bf16.h>

typedef float v8f __attribute__((ext_vector_type(8)));
typedef _Float16 h8  __attribute__((ext_vector_type(8)));
typedef _Float16 h16 __attribute__((ext_vector_type(16)));

#define N_NODES 50000
#define N_EDGES 800000
#define N_GRAPHS 50
#define F_IN 64
#define HID 128
#define LAT 64

#define EPB 256   // edges per block (8 waves x 32 rows)
#define WROWS 320 // 128 (W1) + 128 (W2) + 64 (W3) rows of 128 halves

// XOR-swizzled LDS tile access: rows of 128 halves (256B = 16 chunks of 16B),
// chunk index xored with (row & 15) -> conflict-free for strided-row access.
__device__ inline h16 load_frag_sw(const _Float16* base, int row, int c0, int c1) {
  const _Float16* rp = base + row * HID;
  int r4 = row & 15;
  h8 lo = *(const h8*)(rp + ((c0 ^ r4) << 3));
  h8 hi = *(const h8*)(rp + ((c1 ^ r4) << 3));
  h16 r;
#pragma unroll
  for (int i = 0; i < 8; i++) { r[i] = lo[i]; r[8 + i] = hi[i]; }
  return r;
}
// A 16x32 f16 frag: chunks {ks*4+grp, ks*4+grp+2};  B frag: chunks {ks*4+grp, ks*4+grp+2}

__device__ inline void ln_relu_store(_Float16* shA, v8f* acc, int rbase, int grp, int ln,
                                     const float* gt, const float* bt) {
#pragma unroll
  for (int v = 0; v < 8; v++) {
    float s1 = 0.f, s2 = 0.f;
#pragma unroll
    for (int t = 0; t < 8; t++) { float val = acc[t][v]; s1 += val; s2 += val * val; }
#pragma unroll
    for (int m = 1; m < 16; m <<= 1) { s1 += __shfl_xor(s1, m, 32); s2 += __shfl_xor(s2, m, 32); }
    float mu = s1 * (1.f / 128.f);
    float var = s2 * (1.f / 128.f) - mu * mu;
    float rs = rsqrtf(var + 1e-5f);
    int row = rbase + v + 8 * grp;
    int r4 = row & 15;
#pragma unroll
    for (int t = 0; t < 8; t++) {
      float val = (acc[t][v] - mu) * rs * gt[t] + bt[t];
      int c = t * 16 + ln;
      shA[row * HID + (((c >> 3) ^ r4) << 3) + (c & 7)] = (_Float16)fmaxf(val, 0.f);
    }
  }
}

// ---- prep: x -> f16; last-layer weights -> f16 transposed [N][K=128], W1|W2|W3 contiguous.
// W1 column remap (m_in = [x_i(64) | d2 | xj1 | xj2 | x_j[3:](61)]):
//   k<64 -> row k;  k==64 -> row 125 (d2);  k==65,66 -> 0;  k>=67 -> row k-3
__global__ __launch_bounds__(256) void prep_kernel(
    const float* __restrict__ x, _Float16* __restrict__ xh,
    const float* __restrict__ lW1, const float* __restrict__ lW2, const float* __restrict__ lW3,
    _Float16* __restrict__ wAll)
{
  int i = blockIdx.x * 256 + threadIdx.x;
  if (i < N_NODES * F_IN) xh[i] = (_Float16)x[i];
  if (i < HID * HID) {
    int n = i >> 7, k = i & 127;
    const float* W1l = lW1 + (size_t)2 * 126 * 128;
    float w1;
    if (k < 64)       w1 = W1l[k * 128 + n];
    else if (k == 64) w1 = W1l[125 * 128 + n];
    else if (k < 67)  w1 = 0.f;
    else              w1 = W1l[(k - 3) * 128 + n];
    wAll[i] = (_Float16)w1;
    wAll[HID * HID + i] = (_Float16)lW2[(size_t)2 * 128 * 128 + k * 128 + n];
  }
  if (i < LAT * HID) {
    int n = i >> 7, k = i & 127;           // n in [0,64)
    wAll[2 * HID * HID + i] = (_Float16)lW3[(size_t)2 * 128 * 64 + k * 64 + n];
  }
}

__global__ __launch_bounds__(256) void zero_kernel(
    float* __restrict__ node, float* __restrict__ addp,
    float* __restrict__ maxp, float* __restrict__ cnt)
{
  int i = blockIdx.x * 256 + threadIdx.x;
  if (i < N_NODES * LAT) node[i] = 0.f;
  if (i < N_GRAPHS * LAT) { addp[i] = 0.f; maxp[i] = 0.f; }
  if (i < N_GRAPHS) cnt[i] = 0.f;
}

// ---- main edge-MLP kernel: 256 edges/block, 8 waves x 2 M-stripes, WMMA f16->f32 ----
__global__ __launch_bounds__(256) void edge_mlp_kernel(
    const float* __restrict__ x, const _Float16* __restrict__ xh, const int* __restrict__ ei,
    const _Float16* __restrict__ wAll,
    const float* __restrict__ b1, const float* __restrict__ g1, const float* __restrict__ be1,
    const float* __restrict__ b2, const float* __restrict__ g2, const float* __restrict__ be2,
    const float* __restrict__ b3,
    float* __restrict__ node)
{
  __shared__ __attribute__((aligned(16))) _Float16 shA[EPB * HID];   // 64KB, xor-swizzled rows
  __shared__ __attribute__((aligned(16))) _Float16 shW[WROWS * HID]; // 80KB, xor-swizzled rows
  __shared__ int shDst[EPB];
  const int tid = threadIdx.x;
  const int eb = blockIdx.x * EPB;

  // ---- async-stage all weights into LDS (ASYNCcnt path): 320 rows x 16 chunks of 16B ----
  {
    unsigned ldsbase = (unsigned)(size_t)(&shW[0]);
    unsigned long long gbase = (unsigned long long)wAll;
#pragma unroll
    for (int it = 0; it < 20; it++) {
      int c = it * 256 + tid;              // chunk id in [0, 320*16)
      int row = c >> 4, i = c & 15;
      unsigned loff = ldsbase + (unsigned)(row * 256 + ((i ^ (row & 15)) << 4));
      unsigned goff = (unsigned)(row * 256 + (i << 4));
      asm volatile("global_load_async_to_lds_b128 %0, %1, %2"
                   :: "v"(loff), "v"(goff), "s"(gbase) : "memory");
    }
  }

  // ---- async gather: one edge per thread, x_i -> chunks 0..7, x_j -> chunks 8..15 ----
  int s, d;
  {
    s = ei[eb + tid];            // src -> x_j
    d = ei[N_EDGES + eb + tid];  // dst -> x_i
    shDst[tid] = d;
    unsigned abase = (unsigned)(size_t)(&shA[0]) + (unsigned)(tid * 256);
    unsigned long long gx = (unsigned long long)xh;
    int r4 = tid & 15;
#pragma unroll
    for (int i = 0; i < 8; i++) {
      unsigned loff = abase + (unsigned)((i ^ r4) << 4);
      unsigned goff = (unsigned)(d * 128 + (i << 4));
      asm volatile("global_load_async_to_lds_b128 %0, %1, %2"
                   :: "v"(loff), "v"(goff), "s"(gx) : "memory");
    }
#pragma unroll
    for (int i = 0; i < 8; i++) {
      unsigned loff = abase + (unsigned)(((8 + i) ^ r4) << 4);
      unsigned goff = (unsigned)(s * 128 + (i << 4));
      asm volatile("global_load_async_to_lds_b128 %0, %1, %2"
                   :: "v"(loff), "v"(goff), "s"(gx) : "memory");
    }
  }
  // d2 from the f32 positions
  float d2 = 0.f;
#pragma unroll
  for (int k = 0; k < 3; k++) {
    float df = x[(size_t)s * F_IN + k] - x[(size_t)d * F_IN + k];
    d2 += df * df;
  }
  asm volatile("s_wait_asynccnt 0x0" ::: "memory");
  // overwrite col 64 (chunk 8, half 0) of own row with d2
  shA[tid * HID + ((8 ^ (tid & 15)) << 3)] = (_Float16)d2;
  __syncthreads();   // the ONLY barrier: all activation LDS use below is wave-private

  const int lane = tid & 31;
  const int ln   = lane & 15;
  const int grp  = lane >> 4;
  const int mb   = (tid >> 5) * 32;   // 32 rows per wave (2 stripes)

  v8f acc0[8], acc1[8];
  float gt[8], bt[8];

  // ================= GEMM1: bias in C, then LN+ReLU =================
#pragma unroll
  for (int t = 0; t < 8; t++) {
    int n = t * 16 + ln;
    float bv = b1[n]; gt[t] = g1[n]; bt[t] = be1[n];
    v8f c;
#pragma unroll
    for (int i = 0; i < 8; i++) c[i] = bv;
    acc0[t] = c; acc1[t] = c;
  }
#pragma unroll
  for (int ks = 0; ks < 4; ks++) {
    h16 a0 = load_frag_sw(shA, mb + ln,      ks * 4 + grp, ks * 4 + grp + 2);
    h16 a1 = load_frag_sw(shA, mb + 16 + ln, ks * 4 + grp, ks * 4 + grp + 2);
#pragma unroll
    for (int t = 0; t < 8; t++) {
      h16 b = load_frag_sw(shW, t * 16 + ln, ks * 4 + grp, ks * 4 + grp + 2);
      acc0[t] = __builtin_amdgcn_wmma_f32_16x16x32_f16(false, a0, false, b, (short)0, acc0[t], false, false);
      acc1[t] = __builtin_amdgcn_wmma_f32_16x16x32_f16(false, a1, false, b, (short)0, acc1[t], false, false);
    }
  }
  ln_relu_store(shA, acc0, mb,      grp, ln, gt, bt);
  ln_relu_store(shA, acc1, mb + 16, grp, ln, gt, bt);

  // ================= GEMM2: h1 x W2 (shW rows 128..255), LN+ReLU =================
#pragma unroll
  for (int t = 0; t < 8; t++) {
    int n = t * 16 + ln;
    float bv = b2[n]; gt[t] = g2[n]; bt[t] = be2[n];
    v8f c;
#pragma unroll
    for (int i = 0; i < 8; i++) c[i] = bv;
    acc0[t] = c; acc1[t] = c;
  }
#pragma unroll
  for (int ks = 0; ks < 4; ks++) {
    h16 a0 = load_frag_sw(shA, mb + ln,      ks * 4 + grp, ks * 4 + grp + 2);
    h16 a1 = load_frag_sw(shA, mb + 16 + ln, ks * 4 + grp, ks * 4 + grp + 2);
#pragma unroll
    for (int t = 0; t < 8; t++) {
      h16 b = load_frag_sw(shW, 128 + t * 16 + ln, ks * 4 + grp, ks * 4 + grp + 2);
      acc0[t] = __builtin_amdgcn_wmma_f32_16x16x32_f16(false, a0, false, b, (short)0, acc0[t], false, false);
      acc1[t] = __builtin_amdgcn_wmma_f32_16x16x32_f16(false, a1, false, b, (short)0, acc1[t], false, false);
    }
  }
  ln_relu_store(shA, acc0, mb,      grp, ln, gt, bt);
  ln_relu_store(shA, acc1, mb + 16, grp, ln, gt, bt);

  // ================= GEMM3: h2 x W3 (shW rows 256..319) -> msg, scatter-add =========
  v8f c30[4], c31[4];
#pragma unroll
  for (int t = 0; t < 4; t++) {
    float bv = b3[t * 16 + ln];
    v8f c;
#pragma unroll
    for (int i = 0; i < 8; i++) c[i] = bv;
    c30[t] = c; c31[t] = c;
  }
#pragma unroll
  for (int ks = 0; ks < 4; ks++) {
    h16 a0 = load_frag_sw(shA, mb + ln,      ks * 4 + grp, ks * 4 + grp + 2);
    h16 a1 = load_frag_sw(shA, mb + 16 + ln, ks * 4 + grp, ks * 4 + grp + 2);
#pragma unroll
    for (int t = 0; t < 4; t++) {
      h16 b = load_frag_sw(shW, 256 + t * 16 + ln, ks * 4 + grp, ks * 4 + grp + 2);
      c30[t] = __builtin_amdgcn_wmma_f32_16x16x32_f16(false, a0, false, b, (short)0, c30[t], false, false);
      c31[t] = __builtin_amdgcn_wmma_f32_16x16x32_f16(false, a1, false, b, (short)0, c31[t], false, false);
    }
  }
#pragma unroll
  for (int t = 0; t < 4; t++) {
    int n = t * 16 + ln;
#pragma unroll
    for (int v = 0; v < 8; v++) {
      int row0 = mb + v + 8 * grp;
      atomicAdd(node + (size_t)shDst[row0] * LAT + n, c30[t][v]);
      atomicAdd(node + (size_t)shDst[row0 + 16] * LAT + n, c31[t][v]);
    }
  }
}

// ---- relu + per-graph sum/max/count pooling ----
__global__ __launch_bounds__(256) void relu_pool_kernel(
    const float* __restrict__ node, const int* __restrict__ batch,
    float* __restrict__ addp, float* __restrict__ maxp, float* __restrict__ cnt)
{
  int idx = blockIdx.x * 256 + threadIdx.x;
  if (idx >= N_NODES * LAT) return;
  int n = idx >> 6, f = idx & 63;
  float v = fmaxf(node[idx], 0.f);
  int g = batch[n];
  atomicAdd(addp + g * LAT + f, v);
  atomicMax((int*)(maxp + g * LAT + f), __float_as_int(v));  // v >= 0: int-bit order == float order
  if (f == 0) atomicAdd(cnt + g, 1.0f);
}

// ---- FC head over 50 graphs (tiny, scalar f32, single block) ----
__global__ __launch_bounds__(256) void fc_head_kernel(
    const float* __restrict__ addp, const float* __restrict__ maxp, const float* __restrict__ cnt,
    const float* __restrict__ W1, const float* __restrict__ b1, const float* __restrict__ g1, const float* __restrict__ be1,
    const float* __restrict__ W2, const float* __restrict__ b2, const float* __restrict__ g2, const float* __restrict__ be2,
    const float* __restrict__ W3, const float* __restrict__ b3,
    float* __restrict__ out)
{
  __shared__ float sp[N_GRAPHS * 192];
  __shared__ float sz[N_GRAPHS * 64];
  int tid = threadIdx.x;
  for (int i = tid; i < N_GRAPHS * 64; i += 256) {
    int g = i >> 6, f = i & 63;
    float a = addp[i];
    sp[g * 192 + f] = a;
    sp[g * 192 + 64 + f] = a / cnt[g];
    sp[g * 192 + 128 + f] = maxp[i];
  }
  __syncthreads();
  for (int o = tid; o < N_GRAPHS * 64; o += 256) {
    int r = o >> 6, c = o & 63;
    float acc = b1[c];
    for (int k = 0; k < 192; k++) acc += sp[r * 192 + k] * W1[k * 64 + c];
    sz[o] = acc;
  }
  __syncthreads();
  if (tid < N_GRAPHS) {
    float s1 = 0.f, s2 = 0.f;
    for (int k = 0; k < 64; k++) { float v = sz[tid * 64 + k]; s1 += v; s2 += v * v; }
    float mu = s1 / 64.f, var = s2 / 64.f - mu * mu, rs = rsqrtf(var + 1e-5f);
    for (int k = 0; k < 64; k++) {
      float v = (sz[tid * 64 + k] - mu) * rs * g1[k] + be1[k];
      sz[tid * 64 + k] = fmaxf(v, 0.f);
    }
  }
  __syncthreads();
  for (int o = tid; o < N_GRAPHS * 64; o += 256) {
    int r = o >> 6, c = o & 63;
    float acc = b2[c];
    for (int k = 0; k < 64; k++) acc += sz[r * 64 + k] * W2[k * 64 + c];
    sp[o] = acc;   // reuse sp region
  }
  __syncthreads();
  if (tid < N_GRAPHS) {
    float s1 = 0.f, s2 = 0.f;
    for (int k = 0; k < 64; k++) { float v = sp[tid * 64 + k]; s1 += v; s2 += v * v; }
    float mu = s1 / 64.f, var = s2 / 64.f - mu * mu, rs = rsqrtf(var + 1e-5f);
    for (int k = 0; k < 64; k++) {
      float v = (sp[tid * 64 + k] - mu) * rs * g2[k] + be2[k];
      sp[tid * 64 + k] = fmaxf(v, 0.f);
    }
  }
  __syncthreads();
  if (tid < N_GRAPHS * 4) {
    int r = tid >> 2, c = tid & 3;
    float acc = b3[c];
    for (int k = 0; k < 64; k++) acc += sp[r * 64 + k] * W3[k * 4 + c];
    out[tid] = acc;
  }
}

extern "C" void kernel_launch(void* const* d_in, const int* in_sizes, int n_in,
                              void* d_out, int out_size, void* d_ws, size_t ws_size,
                              hipStream_t stream) {
  const float* x    = (const float*)d_in[0];
  const int*   ei   = (const int*)d_in[1];
  const int*   batch= (const int*)d_in[2];
  const float* lW1  = (const float*)d_in[3];
  const float* lb1  = (const float*)d_in[4];
  const float* lg1  = (const float*)d_in[5];
  const float* lbe1 = (const float*)d_in[6];
  const float* lW2  = (const float*)d_in[7];
  const float* lb2  = (const float*)d_in[8];
  const float* lg2  = (const float*)d_in[9];
  const float* lbe2 = (const float*)d_in[10];
  const float* lW3  = (const float*)d_in[11];
  const float* lb3  = (const float*)d_in[12];
  const float* fW1  = (const float*)d_in[13];
  const float* fb1  = (const float*)d_in[14];
  const float* fg1  = (const float*)d_in[15];
  const float* fbe1 = (const float*)d_in[16];
  const float* fW2  = (const float*)d_in[17];
  const float* fb2  = (const float*)d_in[18];
  const float* fg2  = (const float*)d_in[19];
  const float* fbe2 = (const float*)d_in[20];
  const float* fW3  = (const float*)d_in[21];
  const float* fb3  = (const float*)d_in[22];

  char* ws = (char*)d_ws;
  _Float16* xh   = (_Float16*)ws; ws += (size_t)N_NODES * F_IN * 2;
  _Float16* wAll = (_Float16*)ws; ws += (size_t)WROWS * HID * 2;  // W1|W2|W3 contiguous
  float* node = (float*)ws; ws += (size_t)N_NODES * LAT * 4;
  float* addp = (float*)ws; ws += (size_t)N_GRAPHS * LAT * 4;
  float* maxp = (float*)ws; ws += (size_t)N_GRAPHS * LAT * 4;
  float* cnt  = (float*)ws; ws += (size_t)N_GRAPHS * 4;

  // last-layer (l = N_LAYERS-1 = 2) parameter views
  const float* b1l  = lb1  + 2 * HID;
  const float* g1l  = lg1  + 2 * HID;
  const float* be1l = lbe1 + 2 * HID;
  const float* b2l  = lb2  + 2 * HID;
  const float* g2l  = lg2  + 2 * HID;
  const float* be2l = lbe2 + 2 * HID;
  const float* b3l  = lb3  + 2 * LAT;

  int gprep = (N_NODES * F_IN + 255) / 256;
  prep_kernel<<<gprep, 256, 0, stream>>>(x, xh, lW1, lW2, lW3, wAll);
  int gzero = (N_NODES * LAT + 255) / 256;
  zero_kernel<<<gzero, 256, 0, stream>>>(node, addp, maxp, cnt);
  edge_mlp_kernel<<<N_EDGES / EPB, 256, 0, stream>>>(
      x, xh, ei, wAll, b1l, g1l, be1l, b2l, g2l, be2l, b3l, node);
  relu_pool_kernel<<<(N_NODES * LAT + 255) / 256, 256, 0, stream>>>(node, batch, addp, maxp, cnt);
  fc_head_kernel<<<1, 256, 0, stream>>>(addp, maxp, cnt,
      fW1, fb1, fg1, fbe1, fW2, fb2, fg2, fbe2, fW3, fb3, (float*)d_out);
}
